// BasicBlock_26121991094508
// MI455X (gfx1250) — matrix operand order repaired and checked by reading, not compiled, not run on hardware
//
#include <hip/hip_runtime.h>
#include <hip/hip_bf16.h>

// ---------------------------------------------------------------------------
// Types for CDNA5 WMMA (wave32, bf16 16x16x32 -> f32)
// ---------------------------------------------------------------------------
typedef __attribute__((ext_vector_type(16))) __bf16 v16bf;
typedef __attribute__((ext_vector_type(8)))  __bf16 v8bf;
typedef __attribute__((ext_vector_type(4)))  __bf16 v4bf;
typedef __attribute__((ext_vector_type(8)))  float  v8f;

#define INORM_EPS 1e-5f

static __device__ __forceinline__ v16bf cat8(v8bf lo, v8bf hi) {
  return __builtin_shufflevector(lo, hi, 0,1,2,3,4,5,6,7,8,9,10,11,12,13,14,15);
}

static __device__ __forceinline__ void atomAddF32(float* p, float v) {
  __hip_atomic_fetch_add(p, v, __ATOMIC_RELAXED, __HIP_MEMORY_SCOPE_AGENT);
}

// ---------------------------------------------------------------------------
// Pack W [K,64,64] f32 into the exact per-lane B-matrix VGPR layout (bf16).
// Layout: Wp[k][t(4 col-tiles)][h(2 K-halves)][lane(32)][16 bf16]
//   lane: n = lane&15 (column within tile), kh = lane>>4
//   element j (0..15): K index = h*32 + (j<8 ? kh*8 + j : 16 + kh*8 + j-8), column = t*16 + n
// ---------------------------------------------------------------------------
__global__ void pack_w_kernel(const float* __restrict__ W, __bf16* __restrict__ Wp,
                              int total) {
  int i = blockIdx.x * blockDim.x + threadIdx.x;
  if (i >= total) return;
  int j    = i & 15;
  int lane = (i >> 4) & 31;
  int h    = (i >> 9) & 1;
  int t    = (i >> 10) & 3;
  int k    = i >> 12;
  int n  = lane & 15;
  int kh = lane >> 4;
  int kk = h * 32 + (j < 8 ? kh * 8 + j : 16 + kh * 8 + (j - 8));
  Wp[i] = (__bf16)W[((long)k * 64 + kk) * 64 + t * 16 + n];
}

// ---------------------------------------------------------------------------
// f32 -> bf16 elementwise convert, 4-wide (total must be a multiple of 4)
// ---------------------------------------------------------------------------
__global__ void cvt_bf16_kernel(const float* __restrict__ x, __bf16* __restrict__ xb,
                                long total4) {
  long i = (long)blockIdx.x * blockDim.x + threadIdx.x;
  const long stride = (long)gridDim.x * blockDim.x;
  const float4* x4 = (const float4*)x;
  v4bf* b4 = (v4bf*)xb;
  for (; i < total4; i += stride) {
    float4 v = x4[i];
    v4bf o = { (__bf16)v.x, (__bf16)v.y, (__bf16)v.z, (__bf16)v.w };
    b4[i] = o;
  }
}

// ---------------------------------------------------------------------------
// Zero f32 buffer (float4 stores; n multiple of 4)
// ---------------------------------------------------------------------------
__global__ void zero_f32_kernel(float* __restrict__ p, long n) {
  long i = (long)blockIdx.x * blockDim.x + threadIdx.x;
  const long stride = (long)gridDim.x * blockDim.x;
  long n4 = n >> 2;
  float4* p4 = (float4*)p;
  float4 z = make_float4(0.f, 0.f, 0.f, 0.f);
  for (long q = i; q < n4; q += stride) p4[q] = z;
}

// ---------------------------------------------------------------------------
// Sparse conv: gather 16 rows -> 2x WMMA bf16 per 16-col tile -> f32 atomic
// scatter. 8 independent waves per block; B fragments in VGPRs per wave.
// grid = (ceil(wavesPerK/8), K); no integer division in-kernel.
// ---------------------------------------------------------------------------
__global__ __launch_bounds__(256)
void sconv_wmma_kernel(const __bf16* __restrict__ xb,
                       const int*   __restrict__ in_idx,
                       const int*   __restrict__ out_idx,
                       const __bf16* __restrict__ Wp,
                       float* __restrict__ y,
                       int E, int tilesPerK, int wavesPerK) {
  const int k    = blockIdx.y;
  const int wslt = blockIdx.x * 8 + (threadIdx.x >> 5);
  const int lane = threadIdx.x & 31;
  if (wslt >= wavesPerK) return;          // wave-uniform exit (EXEC stays full)

  const int r  = lane & 15;               // A row / B,D column within 16
  const int hh = lane >> 4;               // K-half selector
  const int c0 = hh * 8;

  // Load B fragments for this k once (4 col-tiles x 2 K-halves = 64 VGPRs).
  v16bf B[4][2];
#pragma unroll
  for (int t = 0; t < 4; ++t) {
#pragma unroll
    for (int h = 0; h < 2; ++h) {
      const v8bf* bp = (const v8bf*)(Wp + ((((long)k * 4 + t) * 2 + h) * 32 + lane) * 16);
      B[t][h] = cat8(bp[0], bp[1]);
    }
  }

  const long kE = (long)k * E;

  for (int tile = wslt; tile < tilesPerK; tile += wavesPerK) {
    const int e0 = tile * 16;
    int e = e0 + r;
    if (e >= E) e = E - 1;                // clamp; invalid rows masked at scatter
    const int ridx = in_idx[kE + e];
    const int oidx = out_idx[kE + e];

    // Gather one 64-channel bf16 row into the 16x32 A layouts (K 0..31, 32..63).
    const __bf16* row = xb + (long)ridx * 64;
    v8bf a0 = *(const v8bf*)(row + c0);
    v8bf a1 = *(const v8bf*)(row + c0 + 16);
    v8bf a2 = *(const v8bf*)(row + c0 + 32);
    v8bf a3 = *(const v8bf*)(row + c0 + 48);
    v16bf A0 = cat8(a0, a1);
    v16bf A1 = cat8(a2, a3);

    // Hoist scatter addresses: D VGPR j holds (row = j + 8*hh, col = r).
    float* baddr[8];
#pragma unroll
    for (int j = 0; j < 8; ++j) {
      const int orow = __builtin_amdgcn_ds_bpermute(((j + hh * 8) & 15) << 2, oidx);
      baddr[j] = y + (long)orow * 64 + r;
    }

    if (e0 + 16 <= E) {                   // fast path: full tile, no guards
#pragma unroll
      for (int t = 0; t < 4; ++t) {
        v8f acc = {};
        acc = __builtin_amdgcn_wmma_f32_16x16x32_bf16(false, A0, false, B[t][0],
                                                      (short)0, acc, false, false);
        acc = __builtin_amdgcn_wmma_f32_16x16x32_bf16(false, A1, false, B[t][1],
                                                      (short)0, acc, false, false);
#pragma unroll
        for (int j = 0; j < 8; ++j) atomAddF32(baddr[j] + t * 16, acc[j]);
      }
    } else {                              // tail tile: per-row guard
#pragma unroll
      for (int t = 0; t < 4; ++t) {
        v8f acc = {};
        acc = __builtin_amdgcn_wmma_f32_16x16x32_bf16(false, A0, false, B[t][0],
                                                      (short)0, acc, false, false);
        acc = __builtin_amdgcn_wmma_f32_16x16x32_bf16(false, A1, false, B[t][1],
                                                      (short)0, acc, false, false);
#pragma unroll
        for (int j = 0; j < 8; ++j) {
          if (e0 + j + hh * 8 < E) atomAddF32(baddr[j] + t * 16, acc[j]);
        }
      }
    }
  }
}

// ---------------------------------------------------------------------------
// Per-channel sum / sumsq over [N,64] -> stats[0..63]=sum, stats[64..127]=sumsq
// blockDim = 256 (stride multiple of 64 keeps channel fixed per thread)
// ---------------------------------------------------------------------------
__global__ __launch_bounds__(256)
void stats_kernel(const float* __restrict__ y, long total, float* __restrict__ stats) {
  const int tid = threadIdx.x;
  float s = 0.f, s2 = 0.f;
  long i = (long)blockIdx.x * blockDim.x + tid;
  const long stride = (long)gridDim.x * blockDim.x;
  for (; i < total; i += stride) {
    float v = y[i];
    s += v;
    s2 += v * v;
  }
  __shared__ float ls[256], ls2[256];
  ls[tid] = s;
  ls2[tid] = s2;
  __syncthreads();
  if (tid < 64) {
    float a  = ls[tid]  + ls[tid + 64]  + ls[tid + 128]  + ls[tid + 192];
    float a2 = ls2[tid] + ls2[tid + 64] + ls2[tid + 128] + ls2[tid + 192];
    atomAddF32(&stats[tid], a);
    atomAddF32(&stats[64 + tid], a2);
  }
}

// stats[128+c] = scale, stats[192+c] = bias   (64 threads)
__global__ void finalize_stats_kernel(float* __restrict__ stats,
                                      const float* __restrict__ gamma,
                                      const float* __restrict__ beta,
                                      float inv_n) {
  int c = threadIdx.x;
  float mu  = stats[c] * inv_n;
  float var = stats[64 + c] * inv_n - mu * mu;
  float sc  = gamma[c] * rsqrtf(var + INORM_EPS);
  stats[128 + c] = sc;
  stats[192 + c] = beta[c] - mu * sc;
}

// relu(inorm(y)) -> bf16 (input to conv2), 4-wide
__global__ void apply_relu_bf16_kernel(const float* __restrict__ y,
                                       const float* __restrict__ stats,
                                       __bf16* __restrict__ zb, long total4) {
  long i = (long)blockIdx.x * blockDim.x + threadIdx.x;
  const long stride = (long)gridDim.x * blockDim.x;
  const float4* y4 = (const float4*)y;
  const float4* sc4 = (const float4*)(stats + 128);
  const float4* bs4 = (const float4*)(stats + 192);
  v4bf* z4 = (v4bf*)zb;
  for (; i < total4; i += stride) {
    int c4 = (int)(i & 15);               // (4*i)&63 / 4
    float4 v = y4[i];
    float4 sc = sc4[c4], bs = bs4[c4];
    v4bf o = { (__bf16)fmaxf(fmaf(v.x, sc.x, bs.x), 0.f),
               (__bf16)fmaxf(fmaf(v.y, sc.y, bs.y), 0.f),
               (__bf16)fmaxf(fmaf(v.z, sc.z, bs.z), 0.f),
               (__bf16)fmaxf(fmaf(v.w, sc.w, bs.w), 0.f) };
    z4[i] = o;
  }
}

// relu(inorm(y) + residual) -> f32 output, 4-wide
__global__ void final_out_kernel(const float* __restrict__ y,
                                 const float* __restrict__ stats,
                                 const float* __restrict__ xres,
                                 float* __restrict__ out, long total4) {
  long i = (long)blockIdx.x * blockDim.x + threadIdx.x;
  const long stride = (long)gridDim.x * blockDim.x;
  const float4* y4 = (const float4*)y;
  const float4* x4 = (const float4*)xres;
  const float4* sc4 = (const float4*)(stats + 128);
  const float4* bs4 = (const float4*)(stats + 192);
  float4* o4 = (float4*)out;
  for (; i < total4; i += stride) {
    int c4 = (int)(i & 15);
    float4 v = y4[i], xr = x4[i];
    float4 sc = sc4[c4], bs = bs4[c4];
    float4 o;
    o.x = fmaxf(fmaf(v.x, sc.x, bs.x) + xr.x, 0.f);
    o.y = fmaxf(fmaf(v.y, sc.y, bs.y) + xr.y, 0.f);
    o.z = fmaxf(fmaf(v.z, sc.z, bs.z) + xr.z, 0.f);
    o.w = fmaxf(fmaf(v.w, sc.w, bs.w) + xr.w, 0.f);
    o4[i] = o;
  }
}

// ---------------------------------------------------------------------------
// Launch
// ---------------------------------------------------------------------------
extern "C" void kernel_launch(void* const* d_in, const int* in_sizes, int n_in,
                              void* d_out, int out_size, void* d_ws, size_t ws_size,
                              hipStream_t stream) {
  const float* x       = (const float*)d_in[0];
  const int*   in_idx  = (const int*)d_in[1];
  const int*   out_idx = (const int*)d_in[2];
  const float* W1      = (const float*)d_in[3];
  const float* W2      = (const float*)d_in[4];
  const float* gamma1  = (const float*)d_in[5];
  const float* beta1   = (const float*)d_in[6];
  const float* gamma2  = (const float*)d_in[7];
  const float* beta2   = (const float*)d_in[8];
  float*       out     = (float*)d_out;

  const int  C  = 64;
  const long NC = in_sizes[0];
  const long N  = NC / C;
  const int  K  = in_sizes[3] / (C * C);
  const int  E  = (int)((long)in_sizes[1] / K);

  // Workspace layout (256B aligned)
  size_t off = 0;
  auto take = [&](size_t bytes) { size_t o = off; off += (bytes + 255) & ~(size_t)255; return o; };
  char* ws = (char*)d_ws;
  __bf16* Wp1    = (__bf16*)(ws + take((size_t)K * 4096 * sizeof(__bf16)));
  __bf16* Wp2    = (__bf16*)(ws + take((size_t)K * 4096 * sizeof(__bf16)));
  float*  stats1 = (float*) (ws + take(256 * sizeof(float)));
  float*  stats2 = (float*) (ws + take(256 * sizeof(float)));
  __bf16* xb     = (__bf16*)(ws + take((size_t)NC * sizeof(__bf16))); // reused as zb
  float*  y      = (float*) (ws + take((size_t)NC * sizeof(float)));  // reused y1/y2
  (void)ws_size; (void)n_in; (void)out_size;

  const int  packTotal  = K * 4096;
  const int  packBlocks = (packTotal + 255) / 256;
  const int  tilesPerK  = (E + 15) / 16;
  const int  wavesPerK  = (tilesPerK + 99) / 100;    // ~100 tiles per wave
  const dim3 convGrid((wavesPerK + 7) / 8, K);
  const int  ewBlocks   = 2048;
  const int  statBlocks = 768;                       // stride multiple of 64
  const long NC4        = NC >> 2;
  const float inv_n     = 1.0f / (float)N;

  // ---- Prep ----
  pack_w_kernel<<<packBlocks, 256, 0, stream>>>(W1, Wp1, packTotal);
  pack_w_kernel<<<packBlocks, 256, 0, stream>>>(W2, Wp2, packTotal);
  cvt_bf16_kernel<<<ewBlocks, 256, 0, stream>>>(x, xb, NC4);
  zero_f32_kernel<<<1, 256, 0, stream>>>(stats1, 256);
  zero_f32_kernel<<<1, 256, 0, stream>>>(stats2, 256);
  zero_f32_kernel<<<ewBlocks, 256, 0, stream>>>(y, NC);

  // ---- Conv1 -> y ----
  sconv_wmma_kernel<<<convGrid, 256, 0, stream>>>(xb, in_idx, out_idx, Wp1, y,
                                                  E, tilesPerK, wavesPerK);
  // ---- InstanceNorm1 + ReLU -> zb (bf16, reuses xb) ----
  stats_kernel<<<statBlocks, 256, 0, stream>>>(y, NC, stats1);
  finalize_stats_kernel<<<1, 64, 0, stream>>>(stats1, gamma1, beta1, inv_n);
  apply_relu_bf16_kernel<<<ewBlocks, 256, 0, stream>>>(y, stats1, xb, NC4);

  // ---- Conv2 -> y (re-zeroed) ----
  zero_f32_kernel<<<ewBlocks, 256, 0, stream>>>(y, NC);
  sconv_wmma_kernel<<<convGrid, 256, 0, stream>>>(xb, in_idx, out_idx, Wp2, y,
                                                  E, tilesPerK, wavesPerK);
  // ---- InstanceNorm2 + residual + ReLU -> out ----
  stats_kernel<<<statBlocks, 256, 0, stream>>>(y, NC, stats2);
  finalize_stats_kernel<<<1, 64, 0, stream>>>(stats2, gamma2, beta2, inv_n);
  final_out_kernel<<<ewBlocks, 256, 0, stream>>>(y, stats2, x, out, NC4);
}
